// S5TrajectoryEncoder_85813446574509
// MI455X (gfx1250) — compile-verified, matrix operand order checked by reading, not run on hardware
//
#include <hip/hip_runtime.h>
#include <math.h>

#define L_SEQ 262144
#define OBS 128
#define H 64
#define P 32
#define NL 3
#define LAT 128
#define CHUNK 1024
#define NCHUNK (L_SEQ / CHUNK)
#define TS 68    // f32 LDS tile row stride (64 + pad)
#define TSB 72   // bf16 LDS tile row stride for 64-wide (144B = 16B aligned)
#define TSB2 136 // bf16 LDS tile row stride for 128-wide (272B = 16B aligned)

typedef __bf16 bf16_t;
typedef bf16_t v16bf __attribute__((ext_vector_type(16)));
typedef float  v8f   __attribute__((ext_vector_type(8)));

// ---------------------------------------------------------------- WMMA helpers
__device__ __forceinline__ v8f wmma_bf16(v16bf a, v16bf b, v8f c) {
  return __builtin_amdgcn_wmma_f32_16x16x32_bf16(false, a, false, b, (short)0, c,
                                                 false, false);
}

// Fragment builder used only in (cold) weight-pack kernels.
// CDNA5 16-bit layout: lane&15 -> row/col, lane>>4 -> K half,
// VGPR v in 0..3 holds pair (2v + 8h), v in 4..7 holds (16 + 2(v-4) + 8h).
template <typename F>
__device__ __forceinline__ v16bf make_frag(int lane, F f) {
  const int half = lane >> 4;
  const int idx  = lane & 15;
  v16bf r;
#pragma unroll
  for (int v = 0; v < 8; ++v) {
    int k0 = ((v < 4) ? (2 * v) : (16 + 2 * (v - 4))) + half * 8;
    r[2 * v]     = (bf16_t)f(idx, k0);
    r[2 * v + 1] = (bf16_t)f(idx, k0 + 1);
  }
  return r;
}

// Hot-path A-fragment: the lane's 16 bf16 values are two contiguous 16B runs.
// Works for LDS or global bf16 row pointers (row-major, 16B-aligned strides).
__device__ __forceinline__ v16bf load_a16(const bf16_t* row, int half, int kk) {
  union { float4 f[2]; v16bf v; } u;
  u.f[0] = *(const float4*)(row + kk + half * 8);
  u.f[1] = *(const float4*)(row + kk + 16 + half * 8);
  return u.v;
}

// ---------------------------------------------------------------- math helpers
__device__ __forceinline__ float gelu_f(float x) {
  const float c = 0.7978845608028654f;
  float t = tanhf(c * (x + 0.044715f * x * x * x));
  return 0.5f * x * (1.0f + t);
}
__device__ __forceinline__ float sigmoid_f(float x) { return 1.0f / (1.0f + expf(-x)); }

// LayerNorm 16x64 (f32 LDS src) -> f32 LDS dst. Wave32-cooperative.
__device__ __forceinline__ void ln_rows_f32(const float (*src)[TS], float (*dst)[TS],
                                            const float* sc, const float* bi, int lane) {
  const float inv = 1.0f / 64.0f;
  for (int m = 0; m < 16; ++m) {
    float v0 = src[m][2 * lane], v1 = src[m][2 * lane + 1];
    float sum = v0 + v1, sq = v0 * v0 + v1 * v1;
#pragma unroll
    for (int o = 16; o; o >>= 1) { sum += __shfl_xor(sum, o, 32); sq += __shfl_xor(sq, o, 32); }
    float mu = sum * inv, var = sq * inv - mu * mu;
    float rs = rsqrtf(var + 1e-6f);
    dst[m][2 * lane]     = (v0 - mu) * rs * sc[2 * lane]     + bi[2 * lane];
    dst[m][2 * lane + 1] = (v1 - mu) * rs * sc[2 * lane + 1] + bi[2 * lane + 1];
  }
}

// LayerNorm 16x64 (f32 LDS src) -> bf16 LDS dst (stride TSB), optional GELU.
template <bool GELU>
__device__ __forceinline__ void ln_rows_bf(const float (*src)[TS], bf16_t* dst,
                                           const float* sc, const float* bi, int lane) {
  const float inv = 1.0f / 64.0f;
  for (int m = 0; m < 16; ++m) {
    float v0 = src[m][2 * lane], v1 = src[m][2 * lane + 1];
    float sum = v0 + v1, sq = v0 * v0 + v1 * v1;
#pragma unroll
    for (int o = 16; o; o >>= 1) { sum += __shfl_xor(sum, o, 32); sq += __shfl_xor(sq, o, 32); }
    float mu = sum * inv, var = sq * inv - mu * mu;
    float rs = rsqrtf(var + 1e-6f);
    float w0 = (v0 - mu) * rs * sc[2 * lane]     + bi[2 * lane];
    float w1 = (v1 - mu) * rs * sc[2 * lane + 1] + bi[2 * lane + 1];
    if (GELU) { w0 = gelu_f(w0); w1 = gelu_f(w1); }
    union { bf16_t b2[2]; unsigned u; } pk;
    pk.b2[0] = (bf16_t)w0; pk.b2[1] = (bf16_t)w1;
    *(unsigned*)&dst[m * TSB + 2 * lane] = pk.u;
  }
}

// Cooperative 16x64 f32 tile load (float4, coalesced).
__device__ __forceinline__ void load_tile16x64(const float* __restrict__ g, int row0,
                                               float (*t)[TS], int lane) {
#pragma unroll
  for (int e = 0; e < 8; ++e) {
    int q = e * 32 + lane, m = q >> 4, c4 = (q & 15) * 4;
    const float4 v = *(const float4*)&g[(size_t)(row0 + m) * H + c4];
    t[m][c4 + 0] = v.x; t[m][c4 + 1] = v.y; t[m][c4 + 2] = v.z; t[m][c4 + 3] = v.w;
  }
}

// =============================================================== weight packing
// Pack W_in (128x64) into B-fragment layout: frag fid = (kk/32)*4 + t, 32B/lane.
__global__ void k_pack_win(const float* __restrict__ Win, bf16_t* __restrict__ dst) {
  int tid = threadIdx.x;              // 512 threads
  int lane = tid & 31, fid = tid >> 5;
  int t = fid & 3, kk = (fid >> 2) * 32;
  v16bf r = make_frag(lane, [&](int nn, int k) {
    return Win[(size_t)(kk + k) * H + t * 16 + nn];
  });
  *(v16bf*)(dst + (size_t)tid * 16) = r;
}

// Per-layer pack: [Bu | C | W1 | W2], each 8 fragments (2 k-steps x 4 n-tiles).
__global__ void k_pack_layer(const float* __restrict__ Lre, const float* __restrict__ Lim,
                             const float* __restrict__ lstep,
                             const float* __restrict__ Bre, const float* __restrict__ Bim,
                             const float* __restrict__ Cre, const float* __restrict__ Cim,
                             const float* __restrict__ W1, const float* __restrict__ W2,
                             bf16_t* __restrict__ dst) {
  int tid = threadIdx.x;              // 256 threads
  int lane = tid & 31, fid = tid >> 5;
  int t = fid & 3, kk = (fid >> 2) * 32;
  int n16 = lane & 15;

  // complex factor f = (Lbar - 1)/Lam for Bu column (t*16+n16) of [Bu_re|Bu_im]
  int cidx = t * 16 + n16, p = cidx & 31, part = cidx >> 5;
  float dl = expf(lstep[p]);
  float lr = Lre[p], li = Lim[p];
  float er = expf(lr * dl);
  float nr = er * cosf(li * dl) - 1.0f;
  float ni = er * sinf(li * dl);
  float d2 = lr * lr + li * li;
  float fre = (nr * lr + ni * li) / d2;
  float fim = (ni * lr - nr * li) / d2;

  v16bf r0 = make_frag(lane, [&](int nn, int k) {
    (void)nn; int kg = kk + k;
    float br = Bre[p * H + kg], bim = Bim[p * H + kg];
    return (part == 0) ? (fre * br - fim * bim) : (fre * bim + fim * br);
  });
  v16bf r1 = make_frag(lane, [&](int nn, int k) {
    int kg = kk + k, n = t * 16 + nn;
    return (kg < P) ? Cre[n * P + kg] : -Cim[n * P + (kg - P)];
  });
  v16bf r2 = make_frag(lane, [&](int nn, int k) {
    return W1[(size_t)(kk + k) * H + t * 16 + nn];
  });
  v16bf r3 = make_frag(lane, [&](int nn, int k) {
    return W2[(size_t)(kk + k) * H + t * 16 + nn];
  });
  size_t o = (size_t)tid * 16;
  *(v16bf*)(dst + o)             = r0;
  *(v16bf*)(dst + 256 * 16 + o)  = r1;
  *(v16bf*)(dst + 512 * 16 + o)  = r2;
  *(v16bf*)(dst + 768 * 16 + o)  = r3;
}

// =============================================================== input projection
__global__ void k_proj_in(const float* __restrict__ x, const bf16_t* __restrict__ winP,
                          const float* __restrict__ bin, float* __restrict__ h) {
  __shared__ bf16_t xa[4][16 * TSB2];
  int wave = threadIdx.x >> 5, lane = threadIdx.x & 31;
  int half = lane >> 4, n16 = lane & 15;
  int row0 = (blockIdx.x * 4 + wave) * 16;
  bf16_t* tx = xa[wave];

  // stage x tile (16x128) as bf16 in LDS
#pragma unroll
  for (int e = 0; e < 16; ++e) {
    int q = e * 32 + lane, m = q >> 5, c4 = (q & 31) * 4;
    float4 f = *(const float4*)&x[(size_t)(row0 + m) * OBS + c4];
    union { bf16_t b4[4]; uint2 u; } pk;
    pk.b4[0] = (bf16_t)f.x; pk.b4[1] = (bf16_t)f.y;
    pk.b4[2] = (bf16_t)f.z; pk.b4[3] = (bf16_t)f.w;
    *(uint2*)&tx[m * TSB2 + c4] = pk.u;
  }
  __syncthreads();

  const v16bf* wp = (const v16bf*)winP;
  const bf16_t* arow = tx + (size_t)n16 * TSB2;
  v8f acc[4] = {};
#pragma unroll
  for (int k2 = 0; k2 < 4; ++k2) {
    v16bf a = load_a16(arow, half, k2 * 32);
#pragma unroll
    for (int t = 0; t < 4; ++t)
      acc[t] = wmma_bf16(a, wp[(k2 * 4 + t) * 32 + lane], acc[t]);
  }
#pragma unroll
  for (int t = 0; t < 4; ++t)
#pragma unroll
    for (int r = 0; r < 8; ++r) {
      int m = r + 8 * half, n = t * 16 + n16;
      float v = acc[t][r] + bin[n];
      h[(size_t)(row0 + m) * H + n] = (v > 0.0f) ? v : 0.01f * v;
    }
}

// =============================================================== layer: LN + Bu
__global__ void k_layer_bu(const float* __restrict__ h,
                           const float* __restrict__ lns, const float* __restrict__ lnb,
                           const bf16_t* __restrict__ packL, float* __restrict__ xs) {
  __shared__ float tf[4][16][TS];
  __shared__ bf16_t tb[4][16 * TSB];
  int wave = threadIdx.x >> 5, lane = threadIdx.x & 31;
  int half = lane >> 4, n16 = lane & 15;
  int row0 = (blockIdx.x * 4 + wave) * 16;

  load_tile16x64(h, row0, tf[wave], lane);
  __syncthreads();
  ln_rows_bf<false>(tf[wave], tb[wave], lns, lnb, lane);
  __syncthreads();

  const v16bf* wp = (const v16bf*)packL;  // Bu pack at offset 0
  const bf16_t* arow = tb[wave] + (size_t)n16 * TSB;
  v8f acc[4] = {};
#pragma unroll
  for (int k2 = 0; k2 < 2; ++k2) {
    v16bf a = load_a16(arow, half, k2 * 32);
#pragma unroll
    for (int q = 0; q < 4; ++q)
      acc[q] = wmma_bf16(a, wp[(k2 * 4 + q) * 32 + lane], acc[q]);
  }
#pragma unroll
  for (int q = 0; q < 4; ++q)
#pragma unroll
    for (int r = 0; r < 8; ++r) {
      int m = r + 8 * half, n = q * 16 + n16;
      xs[((size_t)(row0 + m) * P + (n & 31)) * 2 + (n >> 5)] = acc[q][r];
    }
}

// =============================================================== chunked scan
__device__ __forceinline__ float2 lbar_of(const float* Lre, const float* Lim,
                                          const float* lstep, int p) {
  float dl = expf(lstep[p]);
  float er = expf(Lre[p] * dl);
  return make_float2(er * cosf(Lim[p] * dl), er * sinf(Lim[p] * dl));
}

__global__ void k_scan_local(const float* __restrict__ Lre, const float* __restrict__ Lim,
                             const float* __restrict__ lstep,
                             float* __restrict__ xs, float* __restrict__ csum) {
  int wave = threadIdx.x >> 5, p = threadIdx.x & 31;
  int chunk = blockIdx.x * 4 + wave;
  size_t base = (size_t)chunk * CHUNK;
  float2 lb = lbar_of(Lre, Lim, lstep, p);
  float sr = 0.f, si = 0.f;
  float2* xp = (float2*)xs;
  for (int tt = 0; tt < CHUNK; ++tt) {
    if ((tt & 15) == 0 && tt + 32 < CHUNK)
      __builtin_prefetch(&xp[(base + tt + 32) * P + p], 0, 0);
    float2 u = xp[(base + tt) * P + p];
    float nr = lb.x * sr - lb.y * si + u.x;
    float ni = lb.x * si + lb.y * sr + u.y;
    sr = nr; si = ni;
    xp[(base + tt) * P + p] = make_float2(sr, si);
  }
  ((float2*)csum)[(size_t)chunk * P + p] = make_float2(sr, si);
}

__global__ void k_scan_carry(const float* __restrict__ Lre, const float* __restrict__ Lim,
                             const float* __restrict__ lstep,
                             const float* __restrict__ csum, float* __restrict__ cin) {
  int p = threadIdx.x;
  float dl = expf(lstep[p]);
  float ar = Lre[p] * dl * (float)CHUNK, ai = Lim[p] * dl * (float)CHUNK;
  float er = expf(ar);
  float acr = er * cosf(ai), aci = er * sinf(ai);
  float sr = 0.f, si = 0.f;
  const float2* cs = (const float2*)csum;
  float2* co = (float2*)cin;
  for (int c = 0; c < NCHUNK; ++c) {
    co[(size_t)c * P + p] = make_float2(sr, si);
    float2 bv = cs[(size_t)c * P + p];
    float nr = acr * sr - aci * si + bv.x;
    float ni = acr * si + aci * sr + bv.y;
    sr = nr; si = ni;
  }
}

// Pass 3: apply carry and emit bf16 copy xsb[row][0..31]=re, [32..63]=im.
__global__ void k_scan_apply(const float* __restrict__ Lre, const float* __restrict__ Lim,
                             const float* __restrict__ lstep,
                             float* __restrict__ xs, const float* __restrict__ cin,
                             bf16_t* __restrict__ xsb) {
  int wave = threadIdx.x >> 5, p = threadIdx.x & 31;
  int chunk = blockIdx.x * 4 + wave;
  size_t base = (size_t)chunk * CHUNK;
  float2 lb = lbar_of(Lre, Lim, lstep, p);
  float2 cv = ((const float2*)cin)[(size_t)chunk * P + p];
  float pr = lb.x, pi = lb.y;
  float2* xp = (float2*)xs;
  for (int tt = 0; tt < CHUNK; ++tt) {
    if ((tt & 15) == 0 && tt + 32 < CHUNK)
      __builtin_prefetch(&xp[(base + tt + 32) * P + p], 0, 0);
    float2 u = xp[(base + tt) * P + p];
    u.x += pr * cv.x - pi * cv.y;
    u.y += pr * cv.y + pi * cv.x;
    size_t row = base + tt;
    xsb[row * 64 + p]      = (bf16_t)u.x;
    xsb[row * 64 + 32 + p] = (bf16_t)u.y;
    float npr = pr * lb.x - pi * lb.y;
    float npi = pr * lb.y + pi * lb.x;
    pr = npr; pi = npi;
  }
}

// =============================================================== layer: output mix
__global__ void k_layer_out(float* __restrict__ h, const bf16_t* __restrict__ xsb,
                            const float* __restrict__ lns, const float* __restrict__ lnb,
                            const bf16_t* __restrict__ packL, const float* __restrict__ Dv,
                            const float* __restrict__ b1, const float* __restrict__ b2) {
  __shared__ float thh[4][16][TS];   // residual h
  __shared__ float tbb[4][16][TS];   // xn -> ys
  __shared__ bf16_t tgf[4][16 * TSB];// gelu(LN(ys)) bf16
  int wave = threadIdx.x >> 5, lane = threadIdx.x & 31;
  int half = lane >> 4, n16 = lane & 15;
  int row0 = (blockIdx.x * 4 + wave) * 16;
  float (*th)[TS] = thh[wave];
  float (*tb)[TS] = tbb[wave];
  bf16_t* tg = tgf[wave];

  load_tile16x64(h, row0, th, lane);
  __syncthreads();
  ln_rows_f32(th, tb, lns, lnb, lane);   // tb = xn
  __syncthreads();

  const v16bf* wc  = (const v16bf*)(packL + 256 * 16);
  const v16bf* w1p = (const v16bf*)(packL + 512 * 16);
  const v16bf* w2p = (const v16bf*)(packL + 768 * 16);

  // GEMM1: [xs_re|xs_im] @ [C_re^T; -C_im^T]  (A from global bf16 xsb)
  const bf16_t* arow = xsb + (size_t)(row0 + n16) * 64;
  v8f accY[4] = {};
#pragma unroll
  for (int k2 = 0; k2 < 2; ++k2) {
    v16bf a = load_a16(arow, half, k2 * 32);
#pragma unroll
    for (int t = 0; t < 4; ++t)
      accY[t] = wmma_bf16(a, wc[(k2 * 4 + t) * 32 + lane], accY[t]);
  }
#pragma unroll
  for (int t = 0; t < 4; ++t)
#pragma unroll
    for (int r = 0; r < 8; ++r) {
      int m = r + 8 * half, n = t * 16 + n16;
      tb[m][n] = 2.0f * accY[t][r] + tb[m][n] * Dv[n];   // ys
    }
  __syncthreads();
  ln_rows_bf<true>(tb, tg, lns, lnb, lane);              // tg = gelu(LN(ys))
  __syncthreads();

  // GEMM2/3: gated MLP (A from LDS bf16)
  const bf16_t* grow = tg + (size_t)n16 * TSB;
  v8f g1a[4] = {}, g2a[4] = {};
#pragma unroll
  for (int k2 = 0; k2 < 2; ++k2) {
    v16bf a = load_a16(grow, half, k2 * 32);
#pragma unroll
    for (int t = 0; t < 4; ++t) {
      g1a[t] = wmma_bf16(a, w1p[(k2 * 4 + t) * 32 + lane], g1a[t]);
      g2a[t] = wmma_bf16(a, w2p[(k2 * 4 + t) * 32 + lane], g2a[t]);
    }
  }
#pragma unroll
  for (int t = 0; t < 4; ++t)
#pragma unroll
    for (int r = 0; r < 8; ++r) {
      int m = r + 8 * half, n = t * 16 + n16;
      float a1 = g1a[t][r] + b1[n];
      float a2 = g2a[t][r] + b2[n];
      h[(size_t)(row0 + m) * H + n] = th[m][n] + a1 * sigmoid_f(a2);
    }
}

// =============================================================== pooling + head
__global__ void k_pool_init(float* __restrict__ pooled) {
  if (threadIdx.x < H + 1) pooled[threadIdx.x] = 0.0f;
}

__global__ void k_pool(const float* __restrict__ h, const int* __restrict__ mask,
                       float* __restrict__ pooled) {
  __shared__ float red[256];
  __shared__ float mred[4];
  int tid = threadIdx.x;
  int c = tid & 63, g = tid >> 6;
  size_t base = (size_t)blockIdx.x * 1024;
  float s = 0.f, mc = 0.f;
  for (int j = 0; j < 256; ++j) {
    size_t r = base + g + (size_t)j * 4;
    int mv = mask[r];
    if (mv) s += h[r * H + c];
    if (c == 0) mc += (float)mv;
  }
  red[tid] = s;
  if (c == 0) mred[g] = mc;
  __syncthreads();
  if (tid < 64) {
    float tot = red[tid] + red[tid + 64] + red[tid + 128] + red[tid + 192];
    atomicAdd(&pooled[tid], tot);
  }
  if (tid == 0) atomicAdd(&pooled[64], mred[0] + mred[1] + mred[2] + mred[3]);
}

__global__ void k_final(const float* __restrict__ pooled, const float* __restrict__ Wlat,
                        const float* __restrict__ blat, float* __restrict__ out) {
  int j = threadIdx.x;
  float inv = 1.0f / (pooled[64] + 1e-8f);
  float s = 0.f;
  for (int k = 0; k < H; ++k) s += pooled[k] * Wlat[k * LAT + j];
  out[j] = s * inv + blat[j];
}

// =============================================================== launcher
extern "C" void kernel_launch(void* const* d_in, const int* in_sizes, int n_in,
                              void* d_out, int out_size, void* d_ws, size_t ws_size,
                              hipStream_t stream) {
  (void)in_sizes; (void)n_in; (void)out_size; (void)ws_size;
  const float* x     = (const float*)d_in[0];
  const float* Win   = (const float*)d_in[1];
  const float* bin   = (const float*)d_in[2];
  const float* lnS   = (const float*)d_in[3];
  const float* lnB   = (const float*)d_in[4];
  const float* Lre   = (const float*)d_in[5];
  const float* Lim   = (const float*)d_in[6];
  const float* Bre   = (const float*)d_in[7];
  const float* Bim   = (const float*)d_in[8];
  const float* Cre   = (const float*)d_in[9];
  const float* Cim   = (const float*)d_in[10];
  const float* Dv    = (const float*)d_in[11];
  const float* lstep = (const float*)d_in[12];
  const float* W1    = (const float*)d_in[13];
  const float* b1    = (const float*)d_in[14];
  const float* W2    = (const float*)d_in[15];
  const float* b2    = (const float*)d_in[16];
  const float* Wlat  = (const float*)d_in[17];
  const float* blat  = (const float*)d_in[18];
  const int*   mask  = (const int*)d_in[19];
  float* out = (float*)d_out;

  float*  h      = (float*)d_ws;                         // L*H f32
  float*  xs     = h + (size_t)L_SEQ * H;                // L*P*2 f32
  bf16_t* xsb    = (bf16_t*)(xs + (size_t)L_SEQ * P * 2);// L*64 bf16
  float*  csum   = (float*)(xsb + (size_t)L_SEQ * 64);   // NCHUNK*P*2
  float*  cin    = csum + (size_t)NCHUNK * P * 2;        // NCHUNK*P*2
  float*  pooled = cin + (size_t)NCHUNK * P * 2;         // 72 f32 (65 used)
  bf16_t* winP   = (bf16_t*)(pooled + 72);               // 512*16 bf16
  bf16_t* packL  = winP + (size_t)512 * 16;              // 1024*16 bf16

  dim3 blk(128);
  const int tileBlocks = L_SEQ / 64;  // 4 waves/block, 16 rows/wave

  k_pack_win<<<1, 512, 0, stream>>>(Win, winP);
  k_proj_in<<<tileBlocks, blk, 0, stream>>>(x, winP, bin, h);
  for (int l = 0; l < NL; ++l) {
    k_pack_layer<<<1, 256, 0, stream>>>(Lre + l * P, Lim + l * P, lstep + l * P,
                                        Bre + (size_t)l * P * H, Bim + (size_t)l * P * H,
                                        Cre + (size_t)l * H * P, Cim + (size_t)l * H * P,
                                        W1 + (size_t)l * H * H, W2 + (size_t)l * H * H,
                                        packL);
    k_layer_bu<<<tileBlocks, blk, 0, stream>>>(h, lnS + l * H, lnB + l * H, packL, xs);
    k_scan_local<<<NCHUNK / 4, blk, 0, stream>>>(Lre + l * P, Lim + l * P,
                                                 lstep + l * P, xs, csum);
    k_scan_carry<<<1, 32, 0, stream>>>(Lre + l * P, Lim + l * P, lstep + l * P,
                                       csum, cin);
    k_scan_apply<<<NCHUNK / 4, blk, 0, stream>>>(Lre + l * P, Lim + l * P,
                                                 lstep + l * P, xs, cin, xsb);
    k_layer_out<<<tileBlocks, blk, 0, stream>>>(h, xsb, lnS + l * H, lnB + l * H,
                                                packL, Dv + l * H, b1 + l * H, b2 + l * H);
  }
  k_pool_init<<<1, 128, 0, stream>>>(pooled);
  k_pool<<<256, 256, 0, stream>>>(h, mask, pooled);
  k_final<<<1, LAT, 0, stream>>>(pooled, Wlat, blat, out);
}
